// Criterion1_41059887350011
// MI455X (gfx1250) — compile-verified, matrix operand order checked by reading, not run on hardware
//
#include <hip/hip_runtime.h>
#include <hip/hip_bf16.h>

// ---------------------------------------------------------------------------
// Types for CDNA5 WMMA (wave32, bf16 16x16x32 -> f32)
// ---------------------------------------------------------------------------
typedef __bf16 bf16;
typedef __attribute__((ext_vector_type(16))) __bf16 v16bf;
typedef __attribute__((ext_vector_type(8)))  __bf16 v8bf;
typedef __attribute__((ext_vector_type(8)))  float  v8f;

#define BB   64
#define TT   64
#define WW   20
#define DD   1024
#define TTOT 21
#define EPS  1e-8f
#define MARGIN 0.2f

__device__ __forceinline__ bf16 f2bf(float f) {
    unsigned u = __builtin_bit_cast(unsigned, f);
    u += 0x7FFFu + ((u >> 16) & 1u);           // round-to-nearest-even
    unsigned short h = (unsigned short)(u >> 16);
    return __builtin_bit_cast(bf16, h);
}

// ---------------------------------------------------------------------------
// fp32 -> bf16 convert
// ---------------------------------------------------------------------------
__global__ void cvt_f32_bf16_k(const float* __restrict__ s, bf16* __restrict__ d, int n) {
    int i = blockIdx.x * blockDim.x + threadIdx.x;
    if (i < n) d[i] = f2bf(s[i]);
}

// conv weight (O=1024, C=1024, K=4) -> bf16 (O, K, C)  [K-order = (t_off, c)]
__global__ void perm_conv_w_k(const float* __restrict__ w, bf16* __restrict__ out) {
    int i = blockIdx.x * blockDim.x + threadIdx.x;      // over O*4*1024
    if (i >= 1024 * 4 * 1024) return;
    int c = i & 1023;
    int t = (i >> 10) & 3;
    int o = i >> 12;
    out[i] = f2bf(w[((o << 10) + c) * 4 + t]);
}

// ---------------------------------------------------------------------------
// Generic WMMA GEMM:  Out[m,n] = relu?( sum_k A[m,k]*Bt[n,k] + bias[n] )
// A rows addressed by implicit-im2col map: off(m) = (m/Lm)*sBatch + (m%Lm)*sRow
// Each wave computes a 32x32 tile (2x2 WMMAs); block = 8 waves.
// Requires M%32==0, N%32==0, K%32==0 (true for all calls here).
// ---------------------------------------------------------------------------
__global__ void __launch_bounds__(256) gemm_bf16_wmma_k(
    const bf16* __restrict__ A, const bf16* __restrict__ Bt,
    const float* __restrict__ bias, float* __restrict__ Out,
    int M, int N, int Kd, int Lm, int sRow, int sBatch, int relu)
{
    const int lane  = threadIdx.x & 31;
    const int wave  = threadIdx.x >> 5;
    const int tilesN = N >> 5;
    const int tilesM = M >> 5;
    int tile = blockIdx.x * 8 + wave;
    if (tile >= tilesM * tilesN) return;

    const int mt   = (tile / tilesN) << 1;   // 16-row tile index
    const int nt   = (tile % tilesN) << 1;   // 16-col tile index
    const int half = lane >> 4;
    const int l15  = lane & 15;
    const int hо   = half * 8;

    const int m0 = (mt << 4) + l15;
    const int m1 = m0 + 16;
    const bf16* a0row = A + (size_t)(m0 / Lm) * sBatch + (size_t)(m0 % Lm) * sRow;
    const bf16* a1row = A + (size_t)(m1 / Lm) * sBatch + (size_t)(m1 % Lm) * sRow;
    const bf16* b0row = Bt + (size_t)((nt << 4) + l15) * Kd;
    const bf16* b1row = b0row + (size_t)16 * Kd;

    v8f acc00 = {}, acc01 = {}, acc10 = {}, acc11 = {};

    for (int k = 0; k < Kd; k += 32) {
        if (k + 64 < Kd) {                         // global_prefetch next tiles
            __builtin_prefetch(a0row + k + 64, 0, 1);
            __builtin_prefetch(b0row + k + 64, 0, 1);
        }
        v16bf a0, a1, b0, b1;
        {
            v8bf lo = *(const v8bf*)(a0row + k + hо);
            v8bf hi = *(const v8bf*)(a0row + k + 16 + hо);
            #pragma unroll
            for (int e = 0; e < 8; ++e) { a0[e] = lo[e]; a0[8 + e] = hi[e]; }
        }
        {
            v8bf lo = *(const v8bf*)(a1row + k + hо);
            v8bf hi = *(const v8bf*)(a1row + k + 16 + hо);
            #pragma unroll
            for (int e = 0; e < 8; ++e) { a1[e] = lo[e]; a1[8 + e] = hi[e]; }
        }
        {
            v8bf lo = *(const v8bf*)(b0row + k + hо);
            v8bf hi = *(const v8bf*)(b0row + k + 16 + hо);
            #pragma unroll
            for (int e = 0; e < 8; ++e) { b0[e] = lo[e]; b0[8 + e] = hi[e]; }
        }
        {
            v8bf lo = *(const v8bf*)(b1row + k + hо);
            v8bf hi = *(const v8bf*)(b1row + k + 16 + hо);
            #pragma unroll
            for (int e = 0; e < 8; ++e) { b1[e] = lo[e]; b1[8 + e] = hi[e]; }
        }
        acc00 = __builtin_amdgcn_wmma_f32_16x16x32_bf16(false, a0, false, b0, (short)0, acc00, false, false);
        acc01 = __builtin_amdgcn_wmma_f32_16x16x32_bf16(false, a0, false, b1, (short)0, acc01, false, false);
        acc10 = __builtin_amdgcn_wmma_f32_16x16x32_bf16(false, a1, false, b0, (short)0, acc10, false, false);
        acc11 = __builtin_amdgcn_wmma_f32_16x16x32_bf16(false, a1, false, b1, (short)0, acc11, false, false);
    }

    // Epilogue: C/D layout — lane (l15,half), VGPR r -> (m = half*8+r, n = l15)
    auto store = [&](const v8f& c, int mtile, int ntile) {
        int n = (ntile << 4) + l15;
        float bv = bias ? bias[n] : 0.0f;
        #pragma unroll
        for (int r = 0; r < 8; ++r) {
            int m = (mtile << 4) + hо + r;
            float v = c[r] + bv;
            if (relu) v = fmaxf(v, 0.0f);
            Out[(size_t)m * N + n] = v;
        }
    };
    store(acc00, mt,     nt    );
    store(acc01, mt,     nt + 1);
    store(acc10, mt + 1, nt    );
    store(acc11, mt + 1, nt + 1);
}

// ---------------------------------------------------------------------------
// Per-sentence word Gram matrices: G[i, w, w2] = <words[i,w], words[i,w2]>
// ---------------------------------------------------------------------------
__global__ void __launch_bounds__(256) gram_k(const float* __restrict__ words,
                                              float* __restrict__ G) {
    int i = blockIdx.x;
    const float* wi = words + (size_t)i * WW * DD;
    for (int p = threadIdx.x; p < WW * WW; p += 256) {
        int w = p / WW, w2 = p % WW;
        const float* a = wi + (size_t)w  * DD;
        const float* b = wi + (size_t)w2 * DD;
        float s = 0.f;
        for (int d = 0; d < DD; ++d) s += a[d] * b[d];
        G[i * WW * WW + p] = s;
    }
}

// Row L2 norms: out[row] = sqrt(sum_d X[row,d]^2)
__global__ void __launch_bounds__(128) rownorm_k(const float* __restrict__ X,
                                                 float* __restrict__ out, int D) {
    __shared__ float red[128];
    const float* row = X + (size_t)blockIdx.x * D;
    float s = 0.f;
    for (int d = threadIdx.x; d < D; d += 128) { float v = row[d]; s += v * v; }
    red[threadIdx.x] = s; __syncthreads();
    for (int off = 64; off > 0; off >>= 1) {
        if (threadIdx.x < off) red[threadIdx.x] += red[threadIdx.x + off];
        __syncthreads();
    }
    if (threadIdx.x == 0) out[blockIdx.x] = sqrtf(red[0]);
}

// ---------------------------------------------------------------------------
// Fused attention epilogue: masked softmax over 20 words, cosine sim via
//   v.v_s = sum_w p_w * logit_w ,  |v_s|^2 = p^T G_i p  (no v_s tensor!)
// Block = (i,j) pair, 64 threads; writes scores[i,j] and positive_map rows.
// ---------------------------------------------------------------------------
__global__ void __launch_bounds__(64) sim_scores_k(
    const float* __restrict__ L0, const float* __restrict__ L1, const float* __restrict__ L2,
    const float* __restrict__ vn0, const float* __restrict__ vn1, const float* __restrict__ vn2,
    const float* __restrict__ G, const int* __restrict__ wmask,
    float* __restrict__ scores, float* __restrict__ pm)
{
    __shared__ float g[WW * WW];
    __shared__ float msk[WW];
    __shared__ float red[64];
    const int i = blockIdx.y, j = blockIdx.x;

    for (int p = threadIdx.x; p < WW * WW; p += 64) g[p] = G[i * WW * WW + p];
    if (threadIdx.x < WW) msk[threadIdx.x] = (wmask[i * WW + threadIdx.x] > 0) ? 1.f : 0.f;
    __syncthreads();

    const float* Ls[3]  = { L0, L1, L2 };
    const float* vns[3] = { vn0, vn1, vn2 };
    const int    Tl[3]  = { 16, 4, 1 };
    const int    tb[3]  = { 0, 16, 20 };

    float accum = 0.f;
    #pragma unroll
    for (int l = 0; l < 3; ++l) {
        int t = threadIdx.x;
        if (t < Tl[l]) {
            const float* lr = Ls[l] + (size_t)(j * Tl[l] + t) * (BB * WW) + i * WW;
            float lg[WW]; float mx = -1e30f;
            #pragma unroll
            for (int w = 0; w < WW; ++w) {
                float v = (msk[w] > 0.f) ? lr[w] : -1e30f;
                lg[w] = v; mx = fmaxf(mx, v);
            }
            float p[WW]; float s = 0.f;
            #pragma unroll
            for (int w = 0; w < WW; ++w) {
                float e = (msk[w] > 0.f) ? __expf(lg[w] - mx) : 0.f;
                p[w] = e; s += e;
            }
            float inv = 1.f / fmaxf(s, 1e-30f);
            float numer = 0.f;
            #pragma unroll
            for (int w = 0; w < WW; ++w) numer += (p[w] * inv) * ((msk[w] > 0.f) ? lg[w] : 0.f);
            float q = 0.f;
            for (int w = 0; w < WW; ++w) {
                float pw = p[w] * inv;
                #pragma unroll
                for (int w2 = 0; w2 < WW; ++w2) q += pw * (p[w2] * inv) * g[w * WW + w2];
            }
            float vsn = sqrtf(fmaxf(q, 0.f));
            float vn  = vns[l][j * Tl[l] + t];
            float sim = numer / (fmaxf(vn, EPS) * fmaxf(vsn, EPS));
            accum += sim;
            if (i == j) pm[i * TTOT + tb[l] + t] = sim;
        }
    }
    red[threadIdx.x] = accum; __syncthreads();
    for (int off = 32; off > 0; off >>= 1) {
        if (threadIdx.x < off) red[threadIdx.x] += red[threadIdx.x + off];
        __syncthreads();
    }
    if (threadIdx.x == 0) scores[i * BB + j] = red[0] / (float)TTOT;
}

// ---------------------------------------------------------------------------
// diag() ranking loss over both score matrices; single block.
// g_scores normalized on the fly: gsRaw[i,j] / (max(|s_i|,eps)*max(|gv_j|,eps))
// ---------------------------------------------------------------------------
__global__ void __launch_bounds__(256) loss_k(
    const float* __restrict__ scores, const float* __restrict__ gsRaw,
    const float* __restrict__ sn, const float* __restrict__ gn,
    float* __restrict__ out)
{
    __shared__ float dS[BB], dG[BB], red[256];
    int t = threadIdx.x;
    if (t < BB) {
        dS[t] = scores[t * BB + t];
        dG[t] = gsRaw[t * BB + t] / (fmaxf(sn[t], EPS) * fmaxf(gn[t], EPS));
    }
    __syncthreads();
    float acc = 0.f;
    for (int idx = t; idx < BB * BB; idx += 256) {
        int i = idx >> 6, j = idx & 63;
        if (i != j) {
            float s = scores[idx];
            acc += fmaxf(MARGIN + s - dS[i], 0.f) + fmaxf(MARGIN + s - dS[j], 0.f);
            float g = gsRaw[idx] / (fmaxf(sn[i], EPS) * fmaxf(gn[j], EPS));
            acc += fmaxf(MARGIN + g - dG[i], 0.f) + fmaxf(MARGIN + g - dG[j], 0.f);
        }
    }
    red[t] = acc; __syncthreads();
    for (int off = 128; off > 0; off >>= 1) {
        if (t < off) red[t] += red[t + off];
        __syncthreads();
    }
    if (t == 0) out[0] = red[0] / (float)BB;
}

// ---------------------------------------------------------------------------
// Host launcher
// ---------------------------------------------------------------------------
extern "C" void kernel_launch(void* const* d_in, const int* in_sizes, int n_in,
                              void* d_out, int out_size, void* d_ws, size_t ws_size,
                              hipStream_t stream) {
    const float* video    = (const float*)d_in[0];
    const float* words    = (const float*)d_in[1];
    const int*   w_masks  = (const int*)  d_in[2];
    const float* sent     = (const float*)d_in[3];
    const float* conv0_w  = (const float*)d_in[4];
    const float* conv0_b  = (const float*)d_in[5];
    const float* conv1_w  = (const float*)d_in[6];
    const float* conv1_b  = (const float*)d_in[7];
    const float* conv2_w  = (const float*)d_in[8];
    const float* conv2_b  = (const float*)d_in[9];
    const float* conv1d_w = (const float*)d_in[10];
    const float* conv1d_b = (const float*)d_in[11];
    const float* fc_w     = (const float*)d_in[12];
    const float* fc_b     = (const float*)d_in[13];
    float* out = (float*)d_out;
    float* pm  = out + 1;

    // --- workspace carve-up (256B aligned) ---
    size_t cur = 0;
    auto alloc = [&](size_t bytes) -> char* {
        char* p = (char*)d_ws + cur;
        cur += (bytes + 255) & ~(size_t)255;
        return p;
    };
    const size_t NV = (size_t)BB * TT * DD;       // 4,194,304
    bf16* videoBf = (bf16*)alloc(NV * 2);
    bf16* wt0     = (bf16*)alloc((size_t)1024 * 4096 * 2);
    bf16* wt1     = (bf16*)alloc((size_t)1024 * 4096 * 2);
    bf16* wt2     = (bf16*)alloc((size_t)1024 * 4096 * 2);
    bf16* w1dT    = (bf16*)alloc((size_t)1024 * 1024 * 2);
    bf16* fcT     = (bf16*)alloc((size_t)1024 * 1024 * 2);
    bf16* wordsBf = (bf16*)alloc((size_t)BB * WW * DD * 2);
    bf16* sentBf  = (bf16*)alloc((size_t)BB * DD * 2);
    float* h0 = (float*)alloc((size_t)BB * 16 * DD * 4);  bf16* h0Bf = (bf16*)alloc((size_t)BB * 16 * DD * 2);
    float* h1 = (float*)alloc((size_t)BB *  4 * DD * 4);  bf16* h1Bf = (bf16*)alloc((size_t)BB *  4 * DD * 2);
    float* h2 = (float*)alloc((size_t)BB *  1 * DD * 4);  bf16* h2Bf = (bf16*)alloc((size_t)BB *  1 * DD * 2);
    float* v0 = (float*)alloc((size_t)BB * 16 * DD * 4);  bf16* v0Bf = (bf16*)alloc((size_t)BB * 16 * DD * 2);
    float* v1 = (float*)alloc((size_t)BB *  4 * DD * 4);  bf16* v1Bf = (bf16*)alloc((size_t)BB *  4 * DD * 2);
    float* v2 = (float*)alloc((size_t)BB *  1 * DD * 4);  bf16* v2Bf = (bf16*)alloc((size_t)BB *  1 * DD * 2);
    float* L0 = (float*)alloc((size_t)BB * 16 * BB * WW * 4);
    float* L1 = (float*)alloc((size_t)BB *  4 * BB * WW * 4);
    float* L2 = (float*)alloc((size_t)BB *  1 * BB * WW * 4);
    float* G  = (float*)alloc((size_t)BB * WW * WW * 4);
    float* vn0 = (float*)alloc((size_t)BB * 16 * 4);
    float* vn1 = (float*)alloc((size_t)BB *  4 * 4);
    float* vn2 = (float*)alloc((size_t)BB *  1 * 4);
    float* sn  = (float*)alloc((size_t)BB * 4);
    float* gn  = (float*)alloc((size_t)BB * 4);
    float* gv  = (float*)alloc((size_t)BB * DD * 4);      bf16* gvBf = (bf16*)alloc((size_t)BB * DD * 2);
    float* gsRaw  = (float*)alloc((size_t)BB * BB * 4);
    float* scores = (float*)alloc((size_t)BB * BB * 4);
    (void)ws_size; (void)n_in; (void)in_sizes; (void)out_size;

    auto cvt = [&](const float* s, bf16* d, int n) {
        cvt_f32_bf16_k<<<(n + 255) / 256, 256, 0, stream>>>(s, d, n);
    };
    auto gemm = [&](const bf16* A, const bf16* Bt, const float* bias, float* Out,
                    int M, int N, int Kd, int Lm, int sRow, int sBatch, int relu) {
        int tiles  = (M >> 5) * (N >> 5);
        int blocks = (tiles + 7) / 8;
        gemm_bf16_wmma_k<<<blocks, 256, 0, stream>>>(A, Bt, bias, Out,
                                                     M, N, Kd, Lm, sRow, sBatch, relu);
    };

    // --- operand preparation ---
    cvt(video, videoBf, (int)NV);
    perm_conv_w_k<<<(4194304 + 255) / 256, 256, 0, stream>>>(conv0_w, wt0);
    perm_conv_w_k<<<(4194304 + 255) / 256, 256, 0, stream>>>(conv1_w, wt1);
    perm_conv_w_k<<<(4194304 + 255) / 256, 256, 0, stream>>>(conv2_w, wt2);
    cvt(conv1d_w, w1dT, 1024 * 1024);     // (d,c,1) is already N-major over K=c
    cvt(fc_w,     fcT,  1024 * 1024);     // fc_w[d,c] used as Bt rows
    cvt(words,    wordsBf, BB * WW * DD); // rows n=(i*20+w), len 1024 -> Bt for logits
    cvt(sent,     sentBf,  BB * DD);
    gram_k<<<BB, 256, 0, stream>>>(words, G);

    // --- conv tower (implicit im2col GEMMs, K-order (t_off, c)) ---
    gemm(videoBf, wt0, conv0_b, h0, 1024, 1024, 4096, 16, 4096, 64 * 1024, 1);
    cvt(h0, h0Bf, BB * 16 * DD);
    gemm(h0Bf,   wt1, conv1_b, h1,  256, 1024, 4096,  4, 4096, 16 * 1024, 1);
    cvt(h1, h1Bf, BB * 4 * DD);
    gemm(h1Bf,   wt2, conv2_b, h2,   64, 1024, 4096,  1, 4096,  4 * 1024, 1);
    cvt(h2, h2Bf, BB * DD);

    // --- 1x1 conv per level ---
    gemm(h0Bf, w1dT, conv1d_b, v0, 1024, 1024, 1024, 1024, 1024, 0, 0);
    cvt(v0, v0Bf, BB * 16 * DD);
    gemm(h1Bf, w1dT, conv1d_b, v1,  256, 1024, 1024,  256, 1024, 0, 0);
    cvt(v1, v1Bf, BB * 4 * DD);
    gemm(h2Bf, w1dT, conv1d_b, v2,   64, 1024, 1024,   64, 1024, 0, 0);
    cvt(v2, v2Bf, BB * DD);

    // --- frame norms + attention logits (v x words^T) ---
    rownorm_k<<<BB * 16, 128, 0, stream>>>(v0, vn0, DD);
    rownorm_k<<<BB *  4, 128, 0, stream>>>(v1, vn1, DD);
    rownorm_k<<<BB *  1, 128, 0, stream>>>(v2, vn2, DD);
    gemm(v0Bf, wordsBf, nullptr, L0, 1024, BB * WW, 1024, 1024, 1024, 0, 0);
    gemm(v1Bf, wordsBf, nullptr, L1,  256, BB * WW, 1024,  256, 1024, 0, 0);
    gemm(v2Bf, wordsBf, nullptr, L2,   64, BB * WW, 1024,   64, 1024, 0, 0);

    // --- fused softmax / Gram quadratic-form / cosine / scores ---
    dim3 simGrid(BB, BB);
    sim_scores_k<<<simGrid, 64, 0, stream>>>(L0, L1, L2, vn0, vn1, vn2,
                                             G, w_masks, scores, pm);

    // --- global branch ---
    gemm(h2Bf, fcT, fc_b, gv, 64, 1024, 1024, 64, 1024, 0, 0);
    cvt(gv, gvBf, BB * DD);
    rownorm_k<<<BB, 128, 0, stream>>>(gv, gn, DD);
    rownorm_k<<<BB, 128, 0, stream>>>(sent, sn, DD);
    gemm(sentBf, gvBf, nullptr, gsRaw, 64, 64, 1024, 64, 1024, 0, 0);

    // --- losses ---
    loss_k<<<1, 256, 0, stream>>>(scores, gsRaw, sn, gn, out);
}
